// HeadWiseOrthogonalMultiHeadAttention_39298950759077
// MI455X (gfx1250) — matrix operand order repaired and checked by reading, not compiled, not run on hardware
//
#include <hip/hip_runtime.h>

// ---------------------------------------------------------------------------
// Multi-head attention for MI455X (gfx1250), wave32 + WMMA f16->f32.
//   B=4, S=2048, D=1024, H=16, DK=64
// Pipeline: f32->f16 convert, Q/K projections, V projection with fused
// transpose (Vt[b][dk][s], so attention P*V B-fragments are contiguous),
// fused score+softmax+PV attention kernel, output-projection GEMM (fp32 out).
// All intermediates live in d_ws. Softmax scale folded into Q fragments.
// ---------------------------------------------------------------------------

#define B_ 4
#define S_ 2048
#define D_ 1024
#define H_ 16
#define DK_ 64
#define LDP_ (S_ + 4)   // padded LDS row stride (floats) -> rotates banks

typedef __attribute__((ext_vector_type(16))) _Float16 v16h;
typedef __attribute__((ext_vector_type(8)))  _Float16 v8h;
typedef __attribute__((ext_vector_type(4)))  _Float16 v4h;
typedef __attribute__((ext_vector_type(8)))  float    v8f;
typedef __attribute__((ext_vector_type(4)))  float    v4f;

static __device__ __forceinline__ v8f wmma32(v16h a, v16h b, v8f c) {
  // D = A(16x32 f16) x B(32x16 f16) + C(16x16 f32)
  return __builtin_amdgcn_wmma_f32_16x16x32_f16(false, a, false, b,
                                                (short)0, c, false, false);
}

// A fragment: 16x32 tile at base = &X[m0][k0], row stride ld (f16 elems).
// lane<16 : row=lane,    K = {0..7, 16..23}
// lane>=16: row=lane-16, K = {8..15, 24..31}
static __device__ __forceinline__ v16h load_a16(const _Float16* __restrict__ base,
                                                int ld, int lane) {
  const int row  = lane & 15;
  const int kofs = (lane >> 4) << 3;              // 0 or 8
  const _Float16* p = base + (size_t)row * ld + kofs;
  v8h lo = *(const v8h*)(p);                      // K = kofs+0..7
  v8h hi = *(const v8h*)(p + 16);                 // K = kofs+16..23
  return __builtin_shufflevector(lo, hi, 0,1,2,3,4,5,6,7,8,9,10,11,12,13,14,15);
}

// B fragment: B[k][n] = W[n0+n][k0+k], W row-major [N][K], base=&W[n0][k0].
// lane<16 : col=lane,    K = {0..7, 16..23}
// lane>=16: col=lane-16, K = {8..15, 24..31}
static __device__ __forceinline__ v16h load_b16(const _Float16* __restrict__ base,
                                                int ld, int lane) {
  const int col  = lane & 15;
  const int kofs = (lane >> 4) << 3;              // 0 or 8
  const _Float16* p = base + (size_t)col * ld + kofs;
  v8h lo = *(const v8h*)(p);
  v8h hi = *(const v8h*)(p + 16);
  return __builtin_shufflevector(lo, hi, 0,1,2,3,4,5,6,7,8,9,10,11,12,13,14,15);
}

// A fragment from an fp32 LDS panel (softmax probabilities), cvt to f16.
static __device__ __forceinline__ v16h load_a16_lds_f32(const float* __restrict__ base,
                                                        int ld, int lane) {
  const int row  = lane & 15;
  const int kofs = (lane >> 4) << 3;
  const float* p = base + (size_t)row * ld + kofs;
  v16h r;
#pragma unroll
  for (int e = 0; e < 8; ++e) {
    r[e]     = (_Float16)p[e];
    r[e + 8] = (_Float16)p[e + 16];
  }
  return r;
}

// C/D layout: vgpr r, lane -> row = r + 8*(lane>=16), col = lane&15.
static __device__ __forceinline__ void store_d_f16(const v8f& c,
                                                   _Float16* __restrict__ base,
                                                   int ld, int lane) {
  const int col  = lane & 15;
  const int roff = (lane >> 4) << 3;
#pragma unroll
  for (int r = 0; r < 8; ++r)
    base[(size_t)(roff + r) * ld + col] = (_Float16)c[r];
}

static __device__ __forceinline__ void store_d_f32(const v8f& c,
                                                   float* __restrict__ base,
                                                   int ld, int lane) {
  const int col  = lane & 15;
  const int roff = (lane >> 4) << 3;
#pragma unroll
  for (int r = 0; r < 8; ++r)
    base[(size_t)(roff + r) * ld + col] = c[r];
}

// Transposed f16 store for the V projection: element (m, n) of Y goes to
// Vt[b][n][s] where b = m / S_, s = m % S_ (blocks never straddle batches).
static __device__ __forceinline__ void store_d_f16_vt(const v8f& c,
                                                      _Float16* __restrict__ Y,
                                                      int N, int mbase, int nbase,
                                                      int lane) {
  const int col  = nbase + (lane & 15);
  const int roff = (lane >> 4) << 3;
#pragma unroll
  for (int r = 0; r < 8; ++r) {
    const int m = mbase + roff + r;
    const int b = m >> 11;            // m / S_  (S_ = 2048)
    const int s = m & (S_ - 1);       // m % S_
    Y[((size_t)b * N + col) * S_ + s] = (_Float16)c[r];
  }
}

static __device__ __forceinline__ void store_d_lds(const v8f& c,
                                                   float* base, int ld, int lane) {
  const int col  = lane & 15;
  const int roff = (lane >> 4) << 3;
#pragma unroll
  for (int r = 0; r < 8; ++r)
    base[(size_t)(roff + r) * ld + col] = c[r];
}

// ---------------------------------------------------------------------------
// fp32 -> f16 conversion, 4 elements/thread (b128 load / b64 store)
// ---------------------------------------------------------------------------
__global__ void __launch_bounds__(256)
cvt_f32_f16(const float* __restrict__ in, _Float16* __restrict__ out, int n4) {
  int i = blockIdx.x * 256 + threadIdx.x;
  if (i < n4) {
    v4f a = ((const v4f*)in)[i];
    v4h h;
#pragma unroll
    for (int e = 0; e < 4; ++e) h[e] = (_Float16)a[e];
    ((v4h*)out)[i] = h;
  }
}

// ---------------------------------------------------------------------------
// GEMM: Y[m][n] = sum_k X[m][k] * W[n][k]   (torch F.linear, W row-major [N][K])
// Block = 256 threads = 8 waves (4 along M x 2 along N); wave tile 32x64
// (2 M-tiles x 4 N-tiles -> 8 WMMAs per K-step). Block tile 128x128.
// OUT_MODE: 0 = f16 row-major, 1 = f32 row-major, 2 = f16 V-transposed.
// Grid: (M/128, N/128).
// ---------------------------------------------------------------------------
template <int OUT_MODE>
__global__ void __launch_bounds__(256)
gemm_f16(const _Float16* __restrict__ X, const _Float16* __restrict__ W,
         void* __restrict__ Yv, int M, int N, int K) {
  const int lane = threadIdx.x & 31;
  const int wave = threadIdx.x >> 5;
  const int wm   = wave & 3;          // 4 waves along M
  const int wn   = wave >> 2;         // 2 waves along N
  const int m0   = blockIdx.x * 128 + wm * 32;
  const int n0   = blockIdx.y * 128 + wn * 64;

  v8f acc[2][4] = {};

  for (int k0 = 0; k0 < K; k0 += 32) {
    if (k0 + 32 < K) {  // uniform branch; CDNA5 global_prefetch_b8 hints
      __builtin_prefetch(X + (size_t)(m0 + (lane & 15)) * K + k0 + 32, 0, 0);
      __builtin_prefetch(W + (size_t)(n0 + (lane & 15)) * K + k0 + 32, 0, 0);
    }
    v16h a[2];
#pragma unroll
    for (int i = 0; i < 2; ++i)
      a[i] = load_a16(X + (size_t)(m0 + 16 * i) * K + k0, K, lane);
    v16h b[4];
#pragma unroll
    for (int j = 0; j < 4; ++j)
      b[j] = load_b16(W + (size_t)(n0 + 16 * j) * K + k0, K, lane);
#pragma unroll
    for (int i = 0; i < 2; ++i)
#pragma unroll
      for (int j = 0; j < 4; ++j)
        acc[i][j] = wmma32(a[i], b[j], acc[i][j]);
  }

  if (OUT_MODE == 1) {
    float* Y = (float*)Yv;
#pragma unroll
    for (int i = 0; i < 2; ++i)
#pragma unroll
      for (int j = 0; j < 4; ++j)
        store_d_f32(acc[i][j], Y + (size_t)(m0 + 16 * i) * N + n0 + 16 * j, N, lane);
  } else if (OUT_MODE == 2) {
    _Float16* Y = (_Float16*)Yv;
#pragma unroll
    for (int i = 0; i < 2; ++i)
#pragma unroll
      for (int j = 0; j < 4; ++j)
        store_d_f16_vt(acc[i][j], Y, N, m0 + 16 * i, n0 + 16 * j, lane);
  } else {
    _Float16* Y = (_Float16*)Yv;
#pragma unroll
    for (int i = 0; i < 2; ++i)
#pragma unroll
      for (int j = 0; j < 4; ++j)
        store_d_f16(acc[i][j], Y + (size_t)(m0 + 16 * i) * N + n0 + 16 * j, N, lane);
  }
}

// ---------------------------------------------------------------------------
// Attention: one workgroup (4 waves) per (b, h, 32-query block).
// Phase 1: scores(32 x 2048) = (Q/8) Kt into padded LDS (fp32, ~256.5 KB).
// Phase 2: exact softmax over each row in LDS.
// Phase 3: O(32 x 64) = P x V using Vt[b][dk][s] (contiguous B-fragments),
//          written to [b][s][h*64+dk] layout for the output projection.
// ---------------------------------------------------------------------------
__global__ void __launch_bounds__(128)
attn_kernel(const _Float16* __restrict__ Qh, const _Float16* __restrict__ Kh,
            const _Float16* __restrict__ Vt, _Float16* __restrict__ Oh) {
  extern __shared__ float sc[];                 // [32][LDP_] fp32 scores
  const int lane = threadIdx.x & 31;
  const int wave = threadIdx.x >> 5;
  const int q0   = blockIdx.x * 32;
  const int h    = blockIdx.y;
  const int b    = blockIdx.z;
  const size_t headOff = (size_t)b * S_ * D_ + (size_t)h * DK_;
  const _Float16* Q  = Qh + headOff;            // row stride D_
  const _Float16* Kp = Kh + headOff;
  const _Float16* Vh = Vt + ((size_t)b * D_ + (size_t)h * DK_) * S_;  // [dk][s]

  // ---- Phase 1: scores. Q-fragments hoisted, pre-scaled by 1/sqrt(DK)=0.125
  // (exact power of two -> no precision loss; no inner-loop scaling).
  v16h qa[2][2];
#pragma unroll
  for (int mt = 0; mt < 2; ++mt)
#pragma unroll
    for (int kk = 0; kk < 2; ++kk) {
      v16h f = load_a16(Q + (size_t)(q0 + 16 * mt) * D_ + 32 * kk, D_, lane);
#pragma unroll
      for (int e = 0; e < 16; ++e) f[e] = f[e] * (_Float16)0.125f;
      qa[mt][kk] = f;
    }

  for (int nt = 0; nt < 512; nt += 32) {        // 32 keys / iter per wave
    const int key0 = wave * 512 + nt;
    v8f s[2][2] = {};
#pragma unroll
    for (int j = 0; j < 2; ++j) {
#pragma unroll
      for (int kk = 0; kk < 2; ++kk) {
        v16h kb = load_b16(Kp + (size_t)(key0 + 16 * j) * D_ + 32 * kk, D_, lane);
        s[0][j] = wmma32(qa[0][kk], kb, s[0][j]);
        s[1][j] = wmma32(qa[1][kk], kb, s[1][j]);
      }
      store_d_lds(s[0][j], sc + key0 + 16 * j, LDP_, lane);
      store_d_lds(s[1][j], sc + (size_t)16 * LDP_ + key0 + 16 * j, LDP_, lane);
    }
  }
  __syncthreads();

  // ---- Phase 2: softmax. 4 threads per row; reduce via wave32 shuffles.
  {
    const int row = threadIdx.x >> 2;           // 0..31
    const int sub = threadIdx.x & 3;            // 0..3
    float* r = sc + (size_t)row * LDP_;
    float mx = -3.0e38f;
    for (int c = sub; c < S_; c += 4) mx = fmaxf(mx, r[c]);
    mx = fmaxf(mx, __shfl_xor(mx, 1, 32));
    mx = fmaxf(mx, __shfl_xor(mx, 2, 32));
    float sum = 0.0f;
    for (int c = sub; c < S_; c += 4) {
      float e = __expf(r[c] - mx);
      r[c] = e;
      sum += e;
    }
    sum += __shfl_xor(sum, 1, 32);
    sum += __shfl_xor(sum, 2, 32);
    const float inv = 1.0f / sum;
    for (int c = sub; c < S_; c += 4) r[c] *= inv;
  }
  __syncthreads();

  // ---- Phase 3: O = P x V. Wave w owns dk columns [16w, 16w+16), 2 M-tiles.
  // Vt is row-major [dk][s]  ==  B-layout [N][K] -> contiguous load_b16.
  {
    const int n0 = wave * 16;
    v8f o0 = {}, o1 = {};
    for (int k0 = 0; k0 < S_; k0 += 32) {
      v16h pa0 = load_a16_lds_f32(sc + k0, LDP_, lane);
      v16h pa1 = load_a16_lds_f32(sc + (size_t)16 * LDP_ + k0, LDP_, lane);
      v16h vb  = load_b16(Vh + (size_t)n0 * S_ + k0, S_, lane);
      o0 = wmma32(pa0, vb, o0);
      o1 = wmma32(pa1, vb, o1);
    }
    _Float16* O = Oh + headOff + (size_t)q0 * D_ + n0;   // [b][s][h*64+dk]
    store_d_f16(o0, O, D_, lane);
    store_d_f16(o1, O + (size_t)16 * D_, D_, lane);
  }
}

// ---------------------------------------------------------------------------
// Launch
// ---------------------------------------------------------------------------
extern "C" void kernel_launch(void* const* d_in, const int* in_sizes, int n_in,
                              void* d_out, int out_size, void* d_ws, size_t ws_size,
                              hipStream_t stream) {
  (void)in_sizes; (void)n_in; (void)out_size; (void)ws_size;

  const float* x  = (const float*)d_in[0];
  const float* Wq = (const float*)d_in[1];
  const float* Wk = (const float*)d_in[2];
  const float* Wv = (const float*)d_in[3];
  const float* Wo = (const float*)d_in[4];

  const size_t nx = (size_t)B_ * S_ * D_;   // 8,388,608
  const size_t nw = (size_t)D_ * D_;        // 1,048,576

  _Float16* xh  = (_Float16*)d_ws;
  _Float16* wqh = xh  + nx;
  _Float16* wkh = wqh + nw;
  _Float16* wvh = wkh + nw;
  _Float16* woh = wvh + nw;
  _Float16* qh  = woh + nw;
  _Float16* kh  = qh  + nx;
  _Float16* vt  = kh  + nx;                  // V stored transposed [b][dk][s]
  _Float16* oh  = vt  + nx;                  // total 88 MiB of f16

  // Stage 0: fp32 -> f16 conversions
  cvt_f32_f16<<<(int)(nx / 4 / 256), 256, 0, stream>>>(x,  xh,  (int)(nx / 4));
  cvt_f32_f16<<<(int)(nw / 4 / 256), 256, 0, stream>>>(Wq, wqh, (int)(nw / 4));
  cvt_f32_f16<<<(int)(nw / 4 / 256), 256, 0, stream>>>(Wk, wkh, (int)(nw / 4));
  cvt_f32_f16<<<(int)(nw / 4 / 256), 256, 0, stream>>>(Wv, wvh, (int)(nw / 4));
  cvt_f32_f16<<<(int)(nw / 4 / 256), 256, 0, stream>>>(Wo, woh, (int)(nw / 4));

  // Stage 1: Q/K/V projections (M=8192, N=1024, K=1024), 128x128 block tiles.
  // V projection stores transposed (mode 2) for contiguous attention loads.
  const int M = B_ * S_;
  dim3 ggrid(M / 128, D_ / 128);
  gemm_f16<0><<<ggrid, 256, 0, stream>>>(xh, wqh, qh, M, D_, D_);
  gemm_f16<0><<<ggrid, 256, 0, stream>>>(xh, wkh, kh, M, D_, D_);
  gemm_f16<2><<<ggrid, 256, 0, stream>>>(xh, wvh, vt, M, D_, D_);

  // Stage 2: attention (scores + softmax + PV), ~256.5 KB dynamic LDS
  dim3 agrid(S_ / 32, H_, B_);
  attn_kernel<<<agrid, 128, 32 * LDP_ * sizeof(float), stream>>>(qh, kh, vt, oh);

  // Stage 3: output projection, fp32 result
  gemm_f16<1><<<ggrid, 256, 0, stream>>>(oh, woh, d_out, M, D_, D_);
}
